// GAT_82240033784455
// MI455X (gfx1250) — matrix operand order repaired and checked by reading, not compiled, not run on hardware
//
#include <hip/hip_runtime.h>
#include <hip/hip_bf16.h>
#include <math.h>

// Problem constants (match reference)
#define NNODES 10000
#define NEDGES 160000
#define ET     (NEDGES + NNODES)   // edges + self loops
#define FINF   128
#define HIDF   512
#define NHEAD  8
#define CPH    64
#define NLAYER 3
#define NCLS   10

typedef __attribute__((ext_vector_type(16))) _Float16 v16h;
typedef __attribute__((ext_vector_type(8)))  float    v8f;

__device__ __forceinline__ v8f wmma16(v16h a, v16h b, v8f c) {
  // v_wmma_f32_16x16x32_f16 (codegen-confirmed signature)
  return __builtin_amdgcn_wmma_f32_16x16x32_f16(false, a, false, b, (short)0, c, false, false);
}

// ---------------------------------------------------------------------------
// Weight packing: fp32 [K x N] row-major -> split-f16 fragments.
// Fragment (kt, nt) = 1024 halves: [0,512) = hi, [512,1024) = lo.
// Within each half: lane l (l&15 = column, l>>4 selects K half) holds 16
// halves at lane*16; B-fragment element i corresponds to K = kt*32 +
// 8*(l>>4) + i for i < 8, and kt*32 + 16 + 8*(l>>4) + (i-8) for i >= 8.
// ---------------------------------------------------------------------------
__global__ void k_pack(const float* __restrict__ W, _Float16* __restrict__ pk,
                       int K, int N) {
  int gid    = blockIdx.x * blockDim.x + threadIdx.x;
  int lane   = gid & 31;
  int tile   = gid >> 5;
  int ntiles = N >> 4;
  int ktiles = K >> 5;
  if (tile >= ktiles * ntiles) return;
  int kt = tile / ntiles, nt = tile % ntiles;
  int col   = nt * 16 + (lane & 15);
  int kbase = kt * 32 + ((lane >> 4) << 3);
  size_t base = (size_t)tile * 1024 + lane * 16;
#pragma unroll
  for (int i = 0; i < 16; ++i) {
    float v = W[(size_t)(kbase + (i < 8 ? i : i + 8)) * N + col];
    _Float16 h = (_Float16)v;
    pk[base + i]       = h;                      // hi
    pk[base + 512 + i] = (_Float16)(v - (float)h); // lo residual
  }
}

// ---------------------------------------------------------------------------
// Async copy of one 8KB B-stage (4 fragments, hi+lo) into LDS.
// 256 threads x 2 b128 async transfers; tracked with ASYNCcnt.
// ---------------------------------------------------------------------------
__device__ __forceinline__ void async_copy_stage(const _Float16* __restrict__ gsrc,
                                                 unsigned lds_byte) {
  unsigned t = threadIdx.x;
  unsigned l = lds_byte + t * 16u;
  unsigned long long g = (unsigned long long)(const char*)gsrc + (unsigned long long)t * 16u;
  asm volatile("global_load_async_to_lds_b128 %0, %1, off"
               :: "v"(l), "v"(g) : "memory");
  asm volatile("global_load_async_to_lds_b128 %0, %1, off offset:4096"
               :: "v"(l), "v"(g) : "memory");
}

// ---------------------------------------------------------------------------
// Split-f16 WMMA GEMM with LDS double-buffered B stages.
// C[M,N] = act(A[M,K] @ W[K,N] + bias); one wave -> 16(M) x 64(N) strip,
// 12 WMMAs per k-step (hi*hi + lo*hi + hi*lo), fp32 accumulate.
// All 8 waves of a block share n0 -> B staged once per block in LDS.
// Requires M % 16 == 0 (true here: 10000). act: 0 = none, 1 = ELU.
// ---------------------------------------------------------------------------
__global__ void k_gemm(const float* __restrict__ A, const _Float16* __restrict__ pk,
                       const float* __restrict__ bias, float* __restrict__ C,
                       int M, int N, int K, int act) {
  __shared__ __attribute__((aligned(16))) _Float16 smem[2 * 4096]; // 2 x 8KB stages

  int wave   = threadIdx.x >> 5;
  int lane   = threadIdx.x & 31;
  int mtiles = M >> 4;
  int mtile  = blockIdx.x * (blockDim.x >> 5) + wave;
  bool active = (mtile < mtiles);
  if (!active) mtile = mtiles - 1;        // clamp: keep wave in barriers/copies
  int n0     = blockIdx.y * 64;
  int ntiles = N >> 4;
  int nt0    = n0 >> 4;
  int KT     = K >> 5;

  int m    = mtile * 16 + (lane & 15);    // A-fragment row for this lane
  int koff = (lane >> 4) * 8;             // A-fragment K group offset
  const float* arow = A + (size_t)m * K;

  unsigned sbase = (unsigned)(size_t)(&smem[0]);

  v8f acc[4] = {v8f{}, v8f{}, v8f{}, v8f{}};

  // Prologue: stage kt=0 B into buffer 0; preload kt=0 A floats.
  async_copy_stage(pk + (size_t)nt0 * 1024, sbase);
  float4 a0 = *(const float4*)(arow + koff);
  float4 a1 = *(const float4*)(arow + koff + 4);
  float4 a2 = *(const float4*)(arow + koff + 16);
  float4 a3 = *(const float4*)(arow + koff + 20);

  for (int kt = 0; kt < KT; ++kt) {
    int buf = kt & 1;
    bool more = (kt + 1 < KT);
    // Issue next B stage into the other buffer (safe: end-of-iteration barrier
    // of step kt-1 guaranteed all reads of that buffer completed).
    if (more)
      async_copy_stage(pk + (size_t)((kt + 1) * ntiles + nt0) * 1024,
                       sbase + (unsigned)((buf ^ 1) * 8192));
    // Prefetch next A floats (consumed next iteration; overlaps WMMAs).
    float4 n0f = a0, n1f = a1, n2f = a2, n3f = a3;
    if (more) {
      const float* ar = arow + (kt + 1) * 32 + koff;
      n0f = *(const float4*)(ar);
      n1f = *(const float4*)(ar + 4);
      n2f = *(const float4*)(ar + 16);
      n3f = *(const float4*)(ar + 20);
    }
    // Convert current A floats to split-f16 fragment (ISA 16-bit A layout:
    // elements 0..7 = K koff..koff+7, elements 8..15 = K koff+16..koff+23).
    v16h ahi, alo;
    {
      float f[16] = {a0.x, a0.y, a0.z, a0.w, a1.x, a1.y, a1.z, a1.w,
                     a2.x, a2.y, a2.z, a2.w, a3.x, a3.y, a3.z, a3.w};
#pragma unroll
      for (int i = 0; i < 16; ++i) {
        _Float16 h = (_Float16)f[i];
        ahi[i] = h;
        alo[i] = (_Float16)(f[i] - (float)h);
      }
    }
    // Wait for this step's B stage (newest 2 async ops may stay outstanding),
    // then block-sync so every wave's slice is visible.
    if (more) asm volatile("s_wait_asynccnt 0x2" ::: "memory");
    else      asm volatile("s_wait_asynccnt 0x0" ::: "memory");
    __syncthreads();

    const _Float16* sb = smem + (size_t)buf * 4096;
#pragma unroll
    for (int ns = 0; ns < 4; ++ns) {
      v16h bh = *(const v16h*)(sb + ns * 1024 + lane * 16);
      v16h bl = *(const v16h*)(sb + ns * 1024 + 512 + lane * 16);
      acc[ns] = wmma16(ahi, bh, acc[ns]);
      acc[ns] = wmma16(alo, bh, acc[ns]);
      acc[ns] = wmma16(ahi, bl, acc[ns]);
    }
    __syncthreads();   // all reads of buf done before it is refilled at kt+2
    a0 = n0f; a1 = n1f; a2 = n2f; a3 = n3f;
  }

  if (!active) return;
  // Store: C/D layout -> lane gives col (lane&15), rows r + 8*(lane>=16)
  int rbase = (lane >> 4) * 8;
#pragma unroll
  for (int ns = 0; ns < 4; ++ns) {
    int coln = n0 + ns * 16 + (lane & 15);
    float bv = bias[coln];
#pragma unroll
    for (int r = 0; r < 8; ++r) {
      int row = mtile * 16 + rbase + r;
      float v = acc[ns][r] + bv;
      if (act == 1) v = v > 0.0f ? v : (expf(v) - 1.0f);
      C[(size_t)row * N + coln] = v;
    }
  }
}

// ---------------------------------------------------------------------------
// CSR-by-dst construction (deterministic: rows sorted by original edge id).
// edge_index is [2, E] flattened int32 (row 0 = src, row 1 = dst).
// Self-loops get edge id E + node.
// ---------------------------------------------------------------------------
__global__ void k_initcnt(int* cnt) {
  int i = blockIdx.x * blockDim.x + threadIdx.x;
  if (i < NNODES) cnt[i] = 1;            // self-loop contributes 1
}
__global__ void k_count(const int* __restrict__ ei, int* cnt) {
  int e = blockIdx.x * blockDim.x + threadIdx.x;
  if (e < NEDGES) atomicAdd(&cnt[ei[NEDGES + e]], 1);
}
__global__ void k_scan(const int* __restrict__ cnt, int* __restrict__ row_start, int n) {
  __shared__ int buf[1024];
  __shared__ int carry;
  int t = threadIdx.x;
  if (t == 0) { carry = 0; row_start[0] = 0; }
  __syncthreads();
  for (int base = 0; base < n; base += 1024) {
    int i = base + t;
    buf[t] = (i < n) ? cnt[i] : 0;
    __syncthreads();
    for (int off = 1; off < 1024; off <<= 1) {
      int tmp = (t >= off) ? buf[t - off] : 0;
      __syncthreads();
      buf[t] += tmp;
      __syncthreads();
    }
    int incl = buf[t] + carry;
    if (i < n) row_start[i + 1] = incl;
    __syncthreads();
    if (t == 1023) carry = incl;
    __syncthreads();
  }
}
__global__ void k_self(const int* __restrict__ row_start, int* eids, int* cursor) {
  int i = blockIdx.x * blockDim.x + threadIdx.x;
  if (i < NNODES) { eids[row_start[i]] = NEDGES + i; cursor[i] = 1; }
}
__global__ void k_scatter(const int* __restrict__ ei, const int* __restrict__ row_start,
                          int* cursor, int* eids) {
  int e = blockIdx.x * blockDim.x + threadIdx.x;
  if (e < NEDGES) {
    int d = ei[NEDGES + e];
    int slot = row_start[d] + atomicAdd(&cursor[d], 1);
    eids[slot] = e;
  }
}
__global__ void k_sortrows(const int* __restrict__ row_start, int* eids) {
  int i = blockIdx.x * blockDim.x + threadIdx.x;
  if (i >= NNODES) return;
  int s = row_start[i], e = row_start[i + 1];
  for (int j = s + 1; j < e; ++j) {       // insertion sort (rows avg ~17, L2-resident)
    int key = eids[j], k = j - 1;
    while (k >= s && eids[k] > key) { eids[k + 1] = eids[k]; --k; }
    eids[k + 1] = key;
  }
}
__global__ void k_materialize(const int* __restrict__ eids, const int* __restrict__ ei,
                              const float* __restrict__ dist, int* csrc, float* cdist) {
  int t = blockIdx.x * blockDim.x + threadIdx.x;
  if (t >= ET) return;
  int e = eids[t];
  if (e < NEDGES) { csrc[t] = ei[e]; cdist[t] = dist[e]; }
  else            { csrc[t] = e - NEDGES; cdist[t] = 0.0f; }   // FILL = 0
}

// ---------------------------------------------------------------------------
// Flash-GATv2 aggregation: one wave per destination node, lane l owns
// channels [16l, 16l+16) (head = l/4). Single pass over incoming edges with
// online softmax; no FP atomics; coalesced 2KB/row gathers of xl with
// next-edge prefetch (global_prefetch_b8).
// ---------------------------------------------------------------------------
__global__ void k_gat(const float* __restrict__ xl, const float* __restrict__ xr,
                      const int* __restrict__ row_start, const int* __restrict__ csrc,
                      const float* __restrict__ cdist, const float* __restrict__ We,
                      const float* __restrict__ att, const float* __restrict__ bias,
                      float* __restrict__ out, int applyElu) {
  int wave = threadIdx.x >> 5, lane = threadIdx.x & 31;
  int node = blockIdx.x * (blockDim.x >> 5) + wave;
  if (node >= NNODES) return;
  int ch = lane * 16;

  float xr_r[16], we_r[16], att_r[16], acc[16];
  const float* xrp = xr + (size_t)node * HIDF + ch;
#pragma unroll
  for (int i = 0; i < 16; ++i) {
    xr_r[i] = xrp[i]; we_r[i] = We[ch + i]; att_r[i] = att[ch + i]; acc[i] = 0.0f;
  }
  float runmax = -__builtin_inff(), denom = 0.0f;

  int s = row_start[node], e = row_start[node + 1];
  for (int j = s; j < e; ++j) {
    int   src  = csrc[j];
    float dist = cdist[j];
    // Prefetch next edge's xl row (each lane covers its 64B slice)
    int srcn = csrc[(j + 1 < e) ? (j + 1) : j];
    __builtin_prefetch(xl + (size_t)srcn * HIDF + ch, 0, 0);

    const float* xlp = xl + (size_t)src * HIDF + ch;
    float xls[16];
    float part = 0.0f;
#pragma unroll
    for (int i = 0; i < 16; ++i) {
      float v = xlp[i];
      xls[i] = v;
      float mm = v + xr_r[i] + dist * we_r[i];
      mm = mm > 0.0f ? mm : 0.2f * mm;                 // leaky_relu(0.2)
      part += att_r[i] * mm;
    }
    part += __shfl_xor(part, 1, 32);                   // reduce over head quad
    part += __shfl_xor(part, 2, 32);
    float alpha  = part;
    float newmax = fmaxf(runmax, alpha);
    float scale  = expf(runmax - newmax);              // 0 on first edge (-inf)
    float p      = expf(alpha - newmax);
    denom = denom * scale + p;
#pragma unroll
    for (int i = 0; i < 16; ++i) acc[i] = acc[i] * scale + p * xls[i];
    runmax = newmax;
  }
  float inv = 1.0f / denom;                            // >=1 edge (self loop)
  float* op = out + (size_t)node * HIDF + ch;
#pragma unroll
  for (int i = 0; i < 16; ++i) {
    float v = acc[i] * inv + bias[ch + i];
    if (applyElu) v = v > 0.0f ? v : (expf(v) - 1.0f);
    op[i] = v;
  }
}

// ---------------------------------------------------------------------------
// Classifier: out = log_softmax(elu(h) @ W_after + b_after). One wave/node.
// ---------------------------------------------------------------------------
__global__ void k_logits(const float* __restrict__ h, const float* __restrict__ Wa,
                         const float* __restrict__ ba, float* __restrict__ out) {
  int wave = threadIdx.x >> 5, lane = threadIdx.x & 31;
  int node = blockIdx.x * (blockDim.x >> 5) + wave;
  if (node >= NNODES) return;
  float p[NCLS];
#pragma unroll
  for (int c = 0; c < NCLS; ++c) p[c] = 0.0f;
  const float* hp = h + (size_t)node * HIDF;
  for (int k = lane; k < HIDF; k += 32) {
    float v = hp[k];
    v = v > 0.0f ? v : (expf(v) - 1.0f);               // elu before fcnn_after
    const float* wr = Wa + (size_t)k * NCLS;
#pragma unroll
    for (int c = 0; c < NCLS; ++c) p[c] += v * wr[c];
  }
#pragma unroll
  for (int c = 0; c < NCLS; ++c) {
    p[c] += __shfl_xor(p[c], 1, 32);
    p[c] += __shfl_xor(p[c], 2, 32);
    p[c] += __shfl_xor(p[c], 4, 32);
    p[c] += __shfl_xor(p[c], 8, 32);
    p[c] += __shfl_xor(p[c], 16, 32);
  }
  if (lane == 0) {
    float lg[NCLS], mx = -3.4e38f;
    for (int c = 0; c < NCLS; ++c) { lg[c] = p[c] + ba[c]; mx = fmaxf(mx, lg[c]); }
    float ss = 0.0f;
    for (int c = 0; c < NCLS; ++c) ss += expf(lg[c] - mx);
    float lse = logf(ss) + mx;
    for (int c = 0; c < NCLS; ++c) out[(size_t)node * NCLS + c] = lg[c] - lse;
  }
}

// ---------------------------------------------------------------------------
extern "C" void kernel_launch(void* const* d_in, const int* in_sizes, int n_in,
                              void* d_out, int out_size, void* d_ws, size_t ws_size,
                              hipStream_t stream) {
  (void)in_sizes; (void)n_in; (void)out_size; (void)ws_size;
  const float* x        = (const float*)d_in[0];
  const int*   ei       = (const int*)d_in[1];     // [2,E] int32
  const float* distance = (const float*)d_in[2];
  const float* W_before = (const float*)d_in[3];
  const float* b_before = (const float*)d_in[4];
  const float* Wl       = (const float*)d_in[5];
  const float* bl       = (const float*)d_in[6];
  const float* Wr       = (const float*)d_in[7];
  const float* br       = (const float*)d_in[8];
  const float* We       = (const float*)d_in[9];
  const float* att      = (const float*)d_in[10];
  const float* bias_c   = (const float*)d_in[11];
  const float* W_after  = (const float*)d_in[12];
  const float* b_after  = (const float*)d_in[13];
  float* out = (float*)d_out;

  // Workspace carve-up (~91 MB total)
  char*  ws  = (char*)d_ws;
  size_t off = 0;
  auto alloc = [&](size_t bytes) -> char* {
    char* p = ws + off;
    off = (off + bytes + 255) & ~(size_t)255;
    return p;
  };
  const size_t HN = (size_t)NNODES * HIDF;
  float* h0 = (float*)alloc(HN * 4);
  float* h1 = (float*)alloc(HN * 4);
  float* xl = (float*)alloc(HN * 4);
  float* xr = (float*)alloc(HN * 4);
  // Combined hi|lo fragments: fragment = 1024 halves
  const size_t PACK_BEFORE = (size_t)FINF * HIDF * 2;  // 131072 halves
  const size_t PACK_LAYER  = (size_t)HIDF * HIDF * 2;  // 524288 halves
  const size_t PACK_TOTAL  = PACK_BEFORE + 6 * PACK_LAYER;
  _Float16* pk = (_Float16*)alloc(PACK_TOTAL * 2);
  int*   row_start = (int*)alloc((NNODES + 1) * 4);
  int*   cnt       = (int*)alloc(NNODES * 4);
  int*   cursor    = (int*)alloc(NNODES * 4);
  int*   eids      = (int*)alloc((size_t)ET * 4);
  int*   csrc      = (int*)alloc((size_t)ET * 4);
  float* cdist     = (float*)alloc((size_t)ET * 4);

  // ---- Pack weights into WMMA B fragments (split f16 hi|lo) ----
  {
    int thr = (FINF / 32) * (HIDF / 16) * 32;          // 4096
    k_pack<<<(thr + 255) / 256, 256, 0, stream>>>(W_before, pk, FINF, HIDF);
    int thrL = (HIDF / 32) * (HIDF / 16) * 32;         // 16384
    for (int i = 0; i < NLAYER; ++i) {
      size_t ol  = PACK_BEFORE + (size_t)(2 * i) * PACK_LAYER;
      size_t orr = PACK_BEFORE + (size_t)(2 * i + 1) * PACK_LAYER;
      k_pack<<<(thrL + 255) / 256, 256, 0, stream>>>(Wl + (size_t)i * HIDF * HIDF,
                                                     pk + ol, HIDF, HIDF);
      k_pack<<<(thrL + 255) / 256, 256, 0, stream>>>(Wr + (size_t)i * HIDF * HIDF,
                                                     pk + orr, HIDF, HIDF);
    }
  }

  // ---- Deterministic CSR by destination ----
  k_initcnt<<<(NNODES + 255) / 256, 256, 0, stream>>>(cnt);
  k_count<<<(NEDGES + 255) / 256, 256, 0, stream>>>(ei, cnt);
  k_scan<<<1, 1024, 0, stream>>>(cnt, row_start, NNODES);
  k_self<<<(NNODES + 255) / 256, 256, 0, stream>>>(row_start, eids, cursor);
  k_scatter<<<(NEDGES + 255) / 256, 256, 0, stream>>>(ei, row_start, cursor, eids);
  k_sortrows<<<(NNODES + 255) / 256, 256, 0, stream>>>(row_start, eids);
  k_materialize<<<(ET + 255) / 256, 256, 0, stream>>>(eids, ei, distance, csrc, cdist);

  // ---- h0 = elu(x @ W_before + b_before) ----
  dim3 gblk(256);
  int  mtiles = NNODES / 16;                           // 625
  dim3 ggrid((mtiles + 7) / 8, HIDF / 64);             // 79 x 8
  k_gemm<<<ggrid, gblk, 0, stream>>>(x, pk, b_before, h0, NNODES, HIDF, FINF, 1);

  // ---- GATv2 layers ----
  float* hcur = h0;
  float* hnxt = h1;
  int gatBlocks = (NNODES + 7) / 8;
  for (int i = 0; i < NLAYER; ++i) {
    size_t ol  = PACK_BEFORE + (size_t)(2 * i) * PACK_LAYER;
    size_t orr = PACK_BEFORE + (size_t)(2 * i + 1) * PACK_LAYER;
    k_gemm<<<ggrid, gblk, 0, stream>>>(hcur, pk + ol, bl + (size_t)i * HIDF,
                                       xl, NNODES, HIDF, HIDF, 0);
    k_gemm<<<ggrid, gblk, 0, stream>>>(hcur, pk + orr, br + (size_t)i * HIDF,
                                       xr, NNODES, HIDF, HIDF, 0);
    k_gat<<<gatBlocks, 256, 0, stream>>>(xl, xr, row_start, csrc, cdist,
                                         We + (size_t)i * HIDF, att + (size_t)i * HIDF,
                                         bias_c + (size_t)i * HIDF, hnxt,
                                         (i < NLAYER - 1) ? 1 : 0);
    float* t = hcur; hcur = hnxt; hnxt = t;
  }

  // ---- Classifier + log_softmax ----
  k_logits<<<(NNODES + 7) / 8, 256, 0, stream>>>(hcur, W_after, b_after, out);
}
